// VisionMambaModule_37632503448111
// MI455X (gfx1250) — matrix operand order repaired and checked by reading, not compiled, not run on hardware
//
#include <hip/hip_runtime.h>
#include <cstdint>

typedef __attribute__((ext_vector_type(16))) _Float16 v16h;
typedef __attribute__((ext_vector_type(8)))  _Float16 v8h;
typedef __attribute__((ext_vector_type(8)))  float    v8f;
typedef __attribute__((ext_vector_type(4)))  float    v4f;

static constexpr int Bz = 128;
static constexpr int Cc = 2048;
static constexpr int Hh = 24, Ww = 12;
static constexpr int HID = 768;
static constexpr int INR = 1536;
static constexpr int M9  = Bz * 9;   // 1152 token rows

// ---------------------------------------------------------------------------
// Pool: x (B,C,24,12) -> tok f16 (B*9, C).  l = hi*3+wi, window 8x4 mean.
// 302 MB streaming read -> non-temporal loads (keep L2 for GEMM working set).
// ---------------------------------------------------------------------------
__global__ __launch_bounds__(256) void pool_kernel(const float* __restrict__ x,
                                                   _Float16* __restrict__ tok16) {
  int idx = blockIdx.x * 256 + threadIdx.x;
  const int total = Bz * Cc * 9;
  if (idx >= total) return;
  int l = idx % 9; int bc = idx / 9;
  int c = bc % Cc; int b = bc / Cc;
  int hi = l / 3, wi = l % 3;
  const float* base = x + (size_t)bc * (Hh * Ww) + hi * 8 * Ww + wi * 4;
  float s = 0.f;
#pragma unroll
  for (int r = 0; r < 8; ++r) {
    v4f v = __builtin_nontemporal_load((const v4f*)(base + r * Ww));
    s += v.x + v.y + v.z + v.w;
  }
  tok16[(size_t)(b * 9 + l) * Cc + c] = (_Float16)(s * (1.f / 32.f));
}

// ---------------------------------------------------------------------------
// Weight convert + transpose: src f32 (Ksrc x N) row-major -> dst f16 (N x Kdst)
// row-major, zero-padded rows Ksrc..Kdst-1 (used for dtproj K 48 -> 64).
// ---------------------------------------------------------------------------
__global__ __launch_bounds__(256) void transpose_f16_kernel(const float* __restrict__ src,
                                                            _Float16* __restrict__ dst,
                                                            int Ksrc, int Kdst, int N) {
  int idx = blockIdx.x * 256 + threadIdx.x;
  if (idx >= N * Kdst) return;
  int k = idx % Kdst; int n = idx / Kdst;
  dst[idx] = (_Float16)(k < Ksrc ? src[(size_t)k * N + n] : 0.f);
}

// ---------------------------------------------------------------------------
// WMMA GEMM: C(MxN,f32) = A(MxK,f16 row-major) * Bt(NxK,f16 row-major)^T + bias
// One wave computes a 16 x (16*NT) strip; K stepped by 32 using
// v_wmma_f32_16x16x32_f16.  Fragments are contiguous b128 loads (L2-hot),
// double-buffered with a manual 2x unroll (two named buffer sets -> zero
// register-copy rotation; loads for step s+1 fly under the WMMAs of step s).
// Requires K % 64 == 0 (all call sites satisfy this).
// ---------------------------------------------------------------------------
template <int NT>
__global__ __launch_bounds__(256) void wmma_gemm_nt(const _Float16* __restrict__ A,
                                                    const _Float16* __restrict__ Bt,
                                                    const float* __restrict__ bias,
                                                    float* __restrict__ C,
                                                    int M, int N, int K) {
  int wave = (blockIdx.x * blockDim.x + threadIdx.x) >> 5;
  int lane = threadIdx.x & 31;
  int nCols = N / (16 * NT);
  int mTile = wave / nCols;
  int nCol  = wave % nCols;
  if (mTile * 16 >= M) return;

  int hlf = lane >> 4;        // which half-wave
  int mr  = lane & 15;
  int row = mTile * 16 + mr;  // A row for this lane (both halves share it)
  int nBase = nCol * 16 * NT;

  const _Float16* arow = A + (size_t)row * K + hlf * 8;
  const _Float16* bbase[NT];
#pragma unroll
  for (int t = 0; t < NT; ++t)
    bbase[t] = Bt + (size_t)(nBase + t * 16 + mr) * K + hlf * 8;

  auto loadA = [&](int k0) -> v16h {
    v8h a0 = *(const v8h*)(arow + k0);
    v8h a1 = *(const v8h*)(arow + k0 + 16);
    v16h af;
#pragma unroll
    for (int i = 0; i < 8; ++i) { af[i] = a0[i]; af[i + 8] = a1[i]; }
    return af;
  };
  auto loadB = [&](int t, int k0) -> v16h {
    v8h b0 = *(const v8h*)(bbase[t] + k0);
    v8h b1 = *(const v8h*)(bbase[t] + k0 + 16);
    v16h bf;
#pragma unroll
    for (int i = 0; i < 8; ++i) { bf[i] = b0[i]; bf[i + 8] = b1[i]; }
    return bf;
  };

  v8f acc[NT] = {};
  const int nk = K >> 5;      // even for every call site

  v16h af0 = loadA(0), af1;
  v16h bf0[NT], bf1[NT];
#pragma unroll
  for (int t = 0; t < NT; ++t) bf0[t] = loadB(t, 0);

  int s = 0;
  for (; s + 2 < nk; s += 2) {
    // prefetch step s+1 into buffer 1, multiply step s from buffer 0
    af1 = loadA((s + 1) << 5);
#pragma unroll
    for (int t = 0; t < NT; ++t) bf1[t] = loadB(t, (s + 1) << 5);
#pragma unroll
    for (int t = 0; t < NT; ++t)
      acc[t] = __builtin_amdgcn_wmma_f32_16x16x32_f16(
          false, af0, false, bf0[t], (short)0, acc[t], false, false);
    // prefetch step s+2 into buffer 0, multiply step s+1 from buffer 1
    af0 = loadA((s + 2) << 5);
#pragma unroll
    for (int t = 0; t < NT; ++t) bf0[t] = loadB(t, (s + 2) << 5);
#pragma unroll
    for (int t = 0; t < NT; ++t)
      acc[t] = __builtin_amdgcn_wmma_f32_16x16x32_f16(
          false, af1, false, bf1[t], (short)0, acc[t], false, false);
  }
  // two steps remain: s (already in buffer 0) and s+1
  af1 = loadA((s + 1) << 5);
#pragma unroll
  for (int t = 0; t < NT; ++t) bf1[t] = loadB(t, (s + 1) << 5);
#pragma unroll
  for (int t = 0; t < NT; ++t)
    acc[t] = __builtin_amdgcn_wmma_f32_16x16x32_f16(
        false, af0, false, bf0[t], (short)0, acc[t], false, false);
#pragma unroll
  for (int t = 0; t < NT; ++t)
    acc[t] = __builtin_amdgcn_wmma_f32_16x16x32_f16(
        false, af1, false, bf1[t], (short)0, acc[t], false, false);

  // C/D layout: vgpr r -> row mTile*16 + r + 8*hlf, col nBase + t*16 + mr.
#pragma unroll
  for (int t = 0; t < NT; ++t) {
    int col = nBase + t * 16 + mr;
    float bv = bias ? bias[col] : 0.f;
#pragma unroll
    for (int r = 0; r < 8; ++r) {
      int orow = mTile * 16 + r + hlf * 8;
      C[(size_t)orow * N + col] = acc[t][r] + bv;
    }
  }
}

// ---------------------------------------------------------------------------
// LayerNorm over HID=768 per row (1152 rows); writes f16 for the next GEMM.
// ---------------------------------------------------------------------------
__global__ __launch_bounds__(256) void ln_kernel(const float* __restrict__ in,
                                                 const float* __restrict__ g,
                                                 const float* __restrict__ b,
                                                 _Float16* __restrict__ out) {
  __shared__ float red[256];
  int row = blockIdx.x, tid = threadIdx.x;
  const float* xr = in + (size_t)row * HID;
  float s = 0.f;
  for (int i = tid; i < HID; i += 256) s += xr[i];
  red[tid] = s; __syncthreads();
  for (int o = 128; o > 0; o >>= 1) { if (tid < o) red[tid] += red[tid + o]; __syncthreads(); }
  float mu = red[0] * (1.f / HID);
  __syncthreads();
  float v = 0.f;
  for (int i = tid; i < HID; i += 256) { float d = xr[i] - mu; v += d * d; }
  red[tid] = v; __syncthreads();
  for (int o = 128; o > 0; o >>= 1) { if (tid < o) red[tid] += red[tid + o]; __syncthreads(); }
  float rstd = rsqrtf(red[0] * (1.f / HID) + 1e-5f);
  for (int i = tid; i < HID; i += 256)
    out[(size_t)row * HID + i] = (_Float16)((xr[i] - mu) * rstd * g[i] + b[i]);
}

// ---------------------------------------------------------------------------
// Depthwise conv (k=3, pad 1 over L=9) + SiLU on xs = xz[..., :INR].
// ---------------------------------------------------------------------------
__global__ __launch_bounds__(256) void conv_silu_kernel(const float* __restrict__ xz,
                                                        const float* __restrict__ w,
                                                        const float* __restrict__ cb,
                                                        float* __restrict__ xc32,
                                                        _Float16* __restrict__ xc16) {
  int idx = blockIdx.x * 256 + threadIdx.x;
  const int total = M9 * INR;
  if (idx >= total) return;
  int d = idx % INR; int bl = idx / INR;
  int l = bl % 9; int b = bl / 9;
  float xm = (l > 0) ? xz[(size_t)(b * 9 + l - 1) * (2 * INR) + d] : 0.f;
  float x0 = xz[(size_t)bl * (2 * INR) + d];
  float xp = (l < 8) ? xz[(size_t)(b * 9 + l + 1) * (2 * INR) + d] : 0.f;
  float a = cb[d] + w[d * 3] * xm + w[d * 3 + 1] * x0 + w[d * 3 + 2] * xp;
  float r = a / (1.f + __expf(-a));
  xc32[idx] = r;
  xc16[idx] = (_Float16)r;
}

// d_r (first 48 cols of dbc) -> f16, zero-padded to 64 cols for the K=64 GEMM.
__global__ __launch_bounds__(256) void drpad_kernel(const float* __restrict__ dbc,
                                                    _Float16* __restrict__ drp) {
  int idx = blockIdx.x * 256 + threadIdx.x;
  if (idx >= M9 * 64) return;
  int c = idx % 64; int r = idx / 64;
  drp[idx] = (_Float16)(c < 48 ? dbc[(size_t)r * 80 + c] : 0.f);
}

// ---------------------------------------------------------------------------
// Selective scan: one thread per (b,d); 16 states in registers, L=9 steps.
// ---------------------------------------------------------------------------
__global__ __launch_bounds__(256) void scan_kernel(const float* __restrict__ dpre,
                                                   const float* __restrict__ dbc,
                                                   const float* __restrict__ xz,
                                                   const float* __restrict__ xc32,
                                                   const float* __restrict__ A_log,
                                                   const float* __restrict__ D_param,
                                                   _Float16* __restrict__ y16) {
  int idx = blockIdx.x * 256 + threadIdx.x;
  if (idx >= Bz * INR) return;
  int d = idx % INR; int b = idx / INR;
  float A[16];
#pragma unroll
  for (int n = 0; n < 16; ++n) A[n] = -__expf(A_log[d * 16 + n]);
  float Dv = D_param[d];
  float s[16] = {};
  for (int l = 0; l < 9; ++l) {
    int row = b * 9 + l;
    float dp = dpre[(size_t)row * INR + d];
    float delta = (dp > 20.f) ? dp : log1pf(__expf(dp));
    float u = xc32[(size_t)row * INR + d];
    float du = delta * u;
    float y = 0.f;
#pragma unroll
    for (int n = 0; n < 16; ++n) {
      float Bn = dbc[(size_t)row * 80 + 48 + n];
      float Cn = dbc[(size_t)row * 80 + 64 + n];
      float dA = __expf(delta * A[n]);
      s[n] = dA * s[n] + du * Bn;
      y += s[n] * Cn;
    }
    float zv = xz[(size_t)row * (2 * INR) + INR + d];
    float o = (y + u * Dv) * (zv / (1.f + __expf(-zv)));
    y16[(size_t)row * INR + d] = (_Float16)o;
  }
}

// ---------------------------------------------------------------------------
// Bilinear 3x3 -> 24x12 upsample (half-pixel centers, edge clamp).
// 302 MB streaming write -> non-temporal stores.
// ---------------------------------------------------------------------------
__global__ __launch_bounds__(256) void resize_kernel(const float* __restrict__ o9,
                                                     float* __restrict__ out) {
  long idx = (long)blockIdx.x * 256 + threadIdx.x;
  const long total = (long)Bz * Cc * Hh * Ww;
  if (idx >= total) return;
  int j = (int)(idx % Ww); long t = idx / Ww;
  int i = (int)(t % Hh); t /= Hh;
  int c = (int)(t % Cc); int b = (int)(t / Cc);
  float sy = (i + 0.5f) * (3.f / Hh) - 0.5f;
  float sx = (j + 0.5f) * (3.f / Ww) - 0.5f;
  int iy = (int)floorf(sy); float fy = sy - iy;
  int ix = (int)floorf(sx); float fx = sx - ix;
  int y0 = iy < 0 ? 0 : (iy > 2 ? 2 : iy);
  int y1 = iy + 1 < 0 ? 0 : (iy + 1 > 2 ? 2 : iy + 1);
  int x0 = ix < 0 ? 0 : (ix > 2 ? 2 : ix);
  int x1 = ix + 1 < 0 ? 0 : (ix + 1 > 2 ? 2 : ix + 1);
  const float* p = o9 + (size_t)b * 9 * Cc + c;
  float v00 = p[(size_t)(y0 * 3 + x0) * Cc];
  float v01 = p[(size_t)(y0 * 3 + x1) * Cc];
  float v10 = p[(size_t)(y1 * 3 + x0) * Cc];
  float v11 = p[(size_t)(y1 * 3 + x1) * Cc];
  float v = (1.f - fy) * ((1.f - fx) * v00 + fx * v01) +
            fy * ((1.f - fx) * v10 + fx * v11);
  __builtin_nontemporal_store(v, out + idx);
}

// ---------------------------------------------------------------------------
static inline int cdiv(long a, long b) { return (int)((a + b - 1) / b); }

template <int NT>
static void launch_gemm(const _Float16* A, const _Float16* Bt, const float* bias,
                        float* C, int M, int N, int K, hipStream_t stream) {
  long waves = (long)((M + 15) / 16) * (N / (16 * NT));
  wmma_gemm_nt<NT><<<cdiv(waves * 32, 256), 256, 0, stream>>>(A, Bt, bias, C, M, N, K);
}

extern "C" void kernel_launch(void* const* d_in, const int* in_sizes, int n_in,
                              void* d_out, int out_size, void* d_ws, size_t ws_size,
                              hipStream_t stream) {
  const float* x        = (const float*)d_in[0];
  const float* pe_w     = (const float*)d_in[1];
  const float* pe_b     = (const float*)d_in[2];
  const float* ln_g     = (const float*)d_in[3];
  const float* ln_b     = (const float*)d_in[4];
  const float* in_w     = (const float*)d_in[5];
  const float* conv_w   = (const float*)d_in[6];
  const float* conv_b   = (const float*)d_in[7];
  const float* A_log    = (const float*)d_in[8];
  const float* xproj_w  = (const float*)d_in[9];
  const float* dtproj_w = (const float*)d_in[10];
  const float* dtproj_b = (const float*)d_in[11];
  const float* D_param  = (const float*)d_in[12];
  const float* out_w    = (const float*)d_in[13];
  (void)in_sizes; (void)n_in; (void)out_size; (void)ws_size;

  char* ws = (char*)d_ws; size_t off = 0;
  auto alloc = [&](size_t bytes) -> void* {
    void* p = ws + off; off += (bytes + 255) & ~(size_t)255; return p;
  };
  _Float16* peT   = (_Float16*)alloc((size_t)768 * 2048 * 2);
  _Float16* inT   = (_Float16*)alloc((size_t)3072 * 768 * 2);
  _Float16* xpT   = (_Float16*)alloc((size_t)80 * 1536 * 2);
  _Float16* dtT   = (_Float16*)alloc((size_t)1536 * 64 * 2);
  _Float16* owT   = (_Float16*)alloc((size_t)2048 * 1536 * 2);
  _Float16* tok16 = (_Float16*)alloc((size_t)M9 * 2048 * 2);
  float*    g1    = (float*)   alloc((size_t)M9 * 768 * 4);
  _Float16* h16   = (_Float16*)alloc((size_t)M9 * 768 * 2);
  float*    xzb   = (float*)   alloc((size_t)M9 * 3072 * 4);
  float*    xc32  = (float*)   alloc((size_t)M9 * 1536 * 4);
  _Float16* xc16  = (_Float16*)alloc((size_t)M9 * 1536 * 2);
  float*    dbc   = (float*)   alloc((size_t)M9 * 80 * 4);
  _Float16* drp   = (_Float16*)alloc((size_t)M9 * 64 * 2);
  float*    dpre  = (float*)   alloc((size_t)M9 * 1536 * 4);
  _Float16* y16   = (_Float16*)alloc((size_t)M9 * 1536 * 2);
  float*    o9    = (float*)   alloc((size_t)M9 * 2048 * 4);

  // Weight f32 -> f16 transposes (N x K layout feeds contiguous B fragments).
  transpose_f16_kernel<<<cdiv(768L * 2048, 256), 256, 0, stream>>>(pe_w, peT, 2048, 2048, 768);
  transpose_f16_kernel<<<cdiv(3072L * 768, 256), 256, 0, stream>>>(in_w, inT, 768, 768, 3072);
  transpose_f16_kernel<<<cdiv(80L * 1536, 256), 256, 0, stream>>>(xproj_w, xpT, 1536, 1536, 80);
  transpose_f16_kernel<<<cdiv(1536L * 64, 256), 256, 0, stream>>>(dtproj_w, dtT, 48, 64, 1536);
  transpose_f16_kernel<<<cdiv(2048L * 1536, 256), 256, 0, stream>>>(out_w, owT, 1536, 1536, 2048);

  pool_kernel<<<cdiv((long)Bz * Cc * 9, 256), 256, 0, stream>>>(x, tok16);

  launch_gemm<4>(tok16, peT, pe_b, g1, M9, 768, 2048, stream);          // tok @ pe_w + b
  ln_kernel<<<M9, 256, 0, stream>>>(g1, ln_g, ln_b, h16);               // LayerNorm
  launch_gemm<4>(h16, inT, nullptr, xzb, M9, 3072, 768, stream);        // h @ in_w
  conv_silu_kernel<<<cdiv((long)M9 * INR, 256), 256, 0, stream>>>(xzb, conv_w, conv_b, xc32, xc16);
  launch_gemm<5>(xc16, xpT, nullptr, dbc, M9, 80, 1536, stream);        // xc @ xproj_w
  drpad_kernel<<<cdiv((long)M9 * 64, 256), 256, 0, stream>>>(dbc, drp);
  launch_gemm<4>(drp, dtT, dtproj_b, dpre, M9, 1536, 64, stream);       // d_r @ dtproj_w + b
  scan_kernel<<<cdiv((long)Bz * INR, 256), 256, 0, stream>>>(dpre, dbc, xzb, xc32, A_log, D_param, y16);
  launch_gemm<4>(y16, owT, nullptr, o9, M9, 2048, 1536, stream);        // y @ out_w
  resize_kernel<<<cdiv((long)Bz * Cc * Hh * Ww, 256), 256, 0, stream>>>(o9, (float*)d_out);
}
